// EmotionAttention_31035433681486
// MI455X (gfx1250) — matrix operand order repaired and checked by reading, not compiled, not run on hardware
//
#include <hip/hip_runtime.h>
#include <stdint.h>

typedef __attribute__((ext_vector_type(16))) __bf16 v16bf;
typedef __attribute__((ext_vector_type(8)))  float  v8f;

union Frag { v16bf v; uint4 q[2]; };

__device__ __forceinline__ unsigned short f2bf(float f) {
  unsigned int u = __float_as_uint(f);
  u += 0x7FFFu + ((u >> 16) & 1u);
  return (unsigned short)(u >> 16);
}

__constant__ int c_GL16[16] = {5,3,3,3,3,3,5,5,5,11,3,3,3,3,3,12};
__constant__ int c_GL7[7]   = {5,9,9,25,9,9,12};

// ---------------------------------------------------------------------------
// Kernel 0: build folded weights (bf16, K-major) + BN scale/shift
// ---------------------------------------------------------------------------
__global__ __launch_bounds__(256) void build_weights(
    const float* __restrict__ adj, const float* __restrict__ lgw,
    const float* __restrict__ w1, const float* __restrict__ b1,
    const float* __restrict__ g1, const float* __restrict__ be1,
    const float* __restrict__ m1, const float* __restrict__ v1,
    const float* __restrict__ w2, const float* __restrict__ b2,
    const float* __restrict__ g2, const float* __restrict__ be2,
    const float* __restrict__ m2, const float* __restrict__ v2,
    const float* __restrict__ w3, const float* __restrict__ b3,
    const float* __restrict__ g3, const float* __restrict__ be3,
    const float* __restrict__ m3, const float* __restrict__ v3,
    unsigned short* __restrict__ wbig, unsigned short* __restrict__ w1b,
    unsigned short* __restrict__ w2b,  unsigned short* __restrict__ w3b,
    float* __restrict__ st)
{
  int t = blockIdx.x * blockDim.x + threadIdx.x;
  const int N0 = 320 * 320, N1 = 736 * 256, N2 = 256 * 128, N3 = 128 * 64;
  if (t < N0) {
    int k = t / 320, n = t % 320;
    float val = 0.f;
    if (k < 310 && n < 310) {
      int c = k / 62, j = k % 62, i = n / 5, d = n % 5;
      val = adj[i * 62 + j] * lgw[c * 5 + d];
    }
    wbig[t] = f2bf(val);
    return;
  }
  t -= N0;
  if (t < N1) {
    int k = t / 256, n = t % 256;
    w1b[t] = (k < 735) ? f2bf(w1[k * 256 + n]) : (unsigned short)0;
    return;
  }
  t -= N1;
  if (t < N2) { w2b[t] = f2bf(w2[t]); return; }
  t -= N2;
  if (t < N3) { w3b[t] = f2bf(w3[t]); return; }
  t -= N3;
  if (t < 256) {
    float s = g1[t] * rsqrtf(v1[t] + 1e-5f);
    st[t] = s; st[256 + t] = (b1[t] - m1[t]) * s + be1[t];
    return;
  }
  t -= 256;
  if (t < 128) {
    float s = g2[t] * rsqrtf(v2[t] + 1e-5f);
    st[512 + t] = s; st[640 + t] = (b2[t] - m2[t]) * s + be2[t];
    return;
  }
  t -= 128;
  if (t < 64) {
    float s = g3[t] * rsqrtf(v3[t] + 1e-5f);
    st[768 + t] = s; st[832 + t] = (b3[t] - m3[t]) * s + be3[t];
  }
}

// ---------------------------------------------------------------------------
// Kernel 1: x (f32, B x 310) -> xb (bf16, B x 320, K padded)
// ---------------------------------------------------------------------------
__global__ __launch_bounds__(256) void cvt_x(const float* __restrict__ x,
                                             unsigned short* __restrict__ xb,
                                             int B)
{
  size_t t = (size_t)blockIdx.x * blockDim.x + threadIdx.x;
  if (t >= (size_t)B * 320) return;
  int k = (int)(t % 320);
  size_t b = t / 320;
  xb[t] = (k < 310) ? f2bf(x[b * 310 + k]) : (unsigned short)0;
}

// ---------------------------------------------------------------------------
// Kernel 2: residual = xb @ Wbig + lgcn_b  (WMMA bf16, f32 out)
// 128 threads = 4 waves; each wave owns TWO 16-row M tiles (32 rows), so each
// B fragment feeds two WMMAs (2x arithmetic intensity, 2 indep WMMAs/load).
// ---------------------------------------------------------------------------
__global__ __launch_bounds__(128) void gemm_lgcn(
    const unsigned short* __restrict__ xb, const unsigned short* __restrict__ wb,
    const float* __restrict__ lb, float* __restrict__ resf, int B)
{
  int wave = threadIdx.x >> 5, lane = threadIdx.x & 31;
  int ml = lane & 15, hi = lane >> 4;
  size_t row0 = ((size_t)blockIdx.x * 4 + wave) * 32;

  for (int nt = 0; nt < 20; ++nt) {
    v8f acc0 = {}, acc1 = {};
    for (int kt = 0; kt < 10; ++kt) {
      int k0 = kt * 32 + 8 * hi;
      Frag a0, a1, bq;
      const unsigned short* ap = xb + (row0 + ml) * 320 + k0;
      a0.q[0] = *(const uint4*)ap;
      a0.q[1] = *(const uint4*)(ap + 16);
      a1.q[0] = *(const uint4*)(ap + 16 * 320);
      a1.q[1] = *(const uint4*)(ap + 16 * 320 + 16);
      const unsigned short* bp = wb + (size_t)(kt * 32) * 320 + nt * 16 + ml;
      for (int s = 0; s < 16; ++s)
        ((unsigned short*)&bq)[s] = bp[(size_t)(8 * hi + (s & 7) + 16 * (s >> 3)) * 320];
      acc0 = __builtin_amdgcn_wmma_f32_16x16x32_bf16(
          false, a0.v, false, bq.v, (short)0, acc0, false, false);
      acc1 = __builtin_amdgcn_wmma_f32_16x16x32_bf16(
          false, a1.v, false, bq.v, (short)0, acc1, false, false);
    }
    int n = nt * 16 + ml;
    if (n < 310) {
      float bs = lb[n % 5];
#pragma unroll
      for (int r = 0; r < 8; ++r) {
        int m = r + 8 * hi;
        resf[(row0 + m) * 320 + n]      = acc0[r] + bs;
        resf[(row0 + 16 + m) * 320 + n] = acc1[r] + bs;
      }
    }
  }
}

// ---------------------------------------------------------------------------
// Kernel 3: per-sample graph attention ops -> cat row (bf16) + att1/att2
// one thread per sample; residual staged in dynamic LDS
// ---------------------------------------------------------------------------
__device__ __forceinline__ void meso_block(const float* res, float* scr,
                                           const int* __restrict__ idx,
                                           const float* __restrict__ w,
                                           const int* gl, int ngrp,
                                           unsigned short* out, int stride)
{
  int h = 0;
  for (int grp = 0; grp < ngrp; ++grp) {
    int n = gl[grp];
    float S[10];
#pragma unroll
    for (int q = 0; q < 10; ++q) S[q] = 0.f;
    for (int j = 0; j < n; ++j) {
      int r = idx[h + j];
      float g0 = res[r*5+0], g1 = res[r*5+1], g2 = res[r*5+2];
      float g3 = res[r*5+3], g4 = res[r*5+4];
#pragma unroll
      for (int q = 0; q < 10; ++q)
        S[q] += g0*w[q] + g1*w[10+q] + g2*w[20+q] + g3*w[30+q] + g4*w[40+q];
    }
    float mxv = -3.4e38f;
    for (int j = 0; j < n; ++j) {
      int r = idx[h + j];
      float g0 = res[r*5+0], g1 = res[r*5+1], g2 = res[r*5+2];
      float g3 = res[r*5+3], g4 = res[r*5+4];
      float sc = 0.f;
#pragma unroll
      for (int q = 0; q < 10; ++q) {
        float fw = g0*w[q] + g1*w[10+q] + g2*w[20+q] + g3*w[30+q] + g4*w[40+q];
        sc += fw * S[q];
      }
      scr[j] = sc;
      if (sc > mxv) mxv = sc;
    }
    float sum = 0.f;
    for (int j = 0; j < n; ++j) {
      float e = __expf(scr[j] - mxv);
      scr[j] = e; sum += e;
    }
    float inv = 1.f / sum;
    float o[5] = {0.f, 0.f, 0.f, 0.f, 0.f};
    for (int j = 0; j < n; ++j) {
      int r = idx[h + j];
      float p = scr[j] * inv;
#pragma unroll
      for (int c = 0; c < 5; ++c) o[c] += p * res[r*5+c];
    }
#pragma unroll
    for (int c = 0; c < 5; ++c) out[c * stride + grp] = f2bf(o[c]);
    h += n;
  }
}

__global__ __launch_bounds__(32) void prep_kernel(
    const float* __restrict__ resf,
    const float* __restrict__ ca1w1, const float* __restrict__ ca1w2,
    const float* __restrict__ ca2w1, const float* __restrict__ ca2w2,
    const float* __restrict__ mw1,   const float* __restrict__ mw2,
    const int* __restrict__ idx16,   const int* __restrict__ idx7,
    unsigned short* __restrict__ cat, float* __restrict__ outp, int B)
{
  extern __shared__ float smem[];
  float* res = smem + (size_t)threadIdx.x * 344;   // 310 used + pad
  float* scr = res + 312;                          // 32 softmax slots
  int b = blockIdx.x * 32 + threadIdx.x;

  const float* rrow = resf + (size_t)b * 320;
  for (int k = 0; k < 310; ++k) res[k] = rrow[k];

  unsigned short* crow = cat + (size_t)b * 736;
  // x_flat
  for (int j = 0; j < 62; ++j)
#pragma unroll
    for (int c = 0; c < 5; ++c) crow[c * 62 + j] = f2bf(res[j*5+c]);
  crow[735] = 0;

  // ---- channel attention 1 (over 62) ----
  float avg[5], mx[5];
#pragma unroll
  for (int c = 0; c < 5; ++c) { avg[c] = 0.f; mx[c] = -3.4e38f; }
  for (int j = 0; j < 62; ++j)
#pragma unroll
    for (int c = 0; c < 5; ++c) {
      float v = res[j*5+c];
      avg[c] += v; if (v > mx[c]) mx[c] = v;
    }
#pragma unroll
  for (int c = 0; c < 5; ++c) avg[c] *= (1.f / 62.f);
  float o1[5] = {0.f,0.f,0.f,0.f,0.f};
  for (int k = 0; k < 32; ++k) {
    float ta = 0.f, tm = 0.f;
#pragma unroll
    for (int c = 0; c < 5; ++c) {
      float w = ca1w1[k*5+c];
      ta += avg[c] * w; tm += mx[c] * w;
    }
    float tt = (ta > 0.f ? ta : 0.f) + (tm > 0.f ? tm : 0.f);
#pragma unroll
    for (int c = 0; c < 5; ++c) o1[c] += tt * ca1w2[c*32+k];
  }
  float a1f[5];
  float* att1o = outp + (size_t)B * 64 + (size_t)b * 5;
#pragma unroll
  for (int c = 0; c < 5; ++c) {
    float s = 1.f / (1.f + __expf(-o1[c]));
    att1o[c] = s; a1f[c] = 1.f + s;
  }

  // ---- channel attention 2 (over 5, per row i) ----
  float tA[32], tM[32];
#pragma unroll
  for (int k = 0; k < 32; ++k) { tA[k] = 0.f; tM[k] = 0.f; }
  for (int i = 0; i < 62; ++i) {
    float z0 = res[i*5+0]*a1f[0], z1 = res[i*5+1]*a1f[1], z2 = res[i*5+2]*a1f[2];
    float z3 = res[i*5+3]*a1f[3], z4 = res[i*5+4]*a1f[4];
    float za = (z0+z1+z2+z3+z4) * 0.2f;
    float zm = fmaxf(fmaxf(fmaxf(z0, z1), fmaxf(z2, z3)), z4);
#pragma unroll
    for (int k = 0; k < 32; ++k) {
      float w = ca2w1[k*62+i];
      tA[k] += za * w; tM[k] += zm * w;
    }
  }
#pragma unroll
  for (int k = 0; k < 32; ++k) {
    float a = tA[k], m = tM[k];
    tA[k] = (a > 0.f ? a : 0.f) + (m > 0.f ? m : 0.f);
  }
  float* att2o = outp + (size_t)B * 69 + (size_t)b * 62;
  for (int i = 0; i < 62; ++i) {
    float o = 0.f;
#pragma unroll
    for (int k = 0; k < 32; ++k) o += tA[k] * ca2w2[i*32+k];
    float s = 1.f / (1.f + __expf(-o));
    att2o[i] = s;
    float a2f = 1.f + s;
#pragma unroll
    for (int c = 0; c < 5; ++c)
      crow[310 + c * 62 + i] = f2bf(res[i*5+c] * a1f[c] * a2f);
  }

  // ---- meso blocks ----
  meso_block(res, scr, idx16, mw1, c_GL16, 16, crow + 620, 16);
  meso_block(res, scr, idx7,  mw2, c_GL7,   7, crow + 700, 7);
}

// ---------------------------------------------------------------------------
// Kernel 4: fused 3-layer MLP (WMMA bf16, f32 accum, BN+leakyReLU folded)
// 64 threads = 2 waves; each wave owns a 32-row M block (two 16-row tiles),
// wave-private LDS staging (48 KB static, no barriers needed)
// ---------------------------------------------------------------------------
__global__ __launch_bounds__(64) void mlp3(
    const unsigned short* __restrict__ cat, const unsigned short* __restrict__ w1b,
    const unsigned short* __restrict__ w2b, const unsigned short* __restrict__ w3b,
    const float* __restrict__ st, float* __restrict__ out, int B)
{
  __shared__ unsigned short y1[2][32 * 256];
  __shared__ unsigned short y2[2][32 * 128];
  int wave = threadIdx.x >> 5, lane = threadIdx.x & 31;
  int ml = lane & 15, hi = lane >> 4;
  size_t row0 = ((size_t)blockIdx.x * 2 + wave) * 32;
  const float* s1 = st;        const float* t1 = st + 256;
  const float* s2 = st + 512;  const float* t2 = st + 640;
  const float* s3 = st + 768;  const float* t3 = st + 832;

  // layer 1: 736 -> 256
  for (int nt = 0; nt < 16; ++nt) {
    v8f acc0 = {}, acc1 = {};
    for (int kt = 0; kt < 23; ++kt) {
      int k0 = kt * 32 + 8 * hi;
      Frag a0, a1, bq;
      const unsigned short* ap = cat + (row0 + ml) * 736 + k0;
      a0.q[0] = *(const uint4*)ap;
      a0.q[1] = *(const uint4*)(ap + 16);
      a1.q[0] = *(const uint4*)(ap + 16 * 736);
      a1.q[1] = *(const uint4*)(ap + 16 * 736 + 16);
      const unsigned short* bp = w1b + (size_t)(kt * 32) * 256 + nt * 16 + ml;
      for (int s = 0; s < 16; ++s)
        ((unsigned short*)&bq)[s] = bp[(size_t)(8 * hi + (s & 7) + 16 * (s >> 3)) * 256];
      acc0 = __builtin_amdgcn_wmma_f32_16x16x32_bf16(
          false, a0.v, false, bq.v, (short)0, acc0, false, false);
      acc1 = __builtin_amdgcn_wmma_f32_16x16x32_bf16(
          false, a1.v, false, bq.v, (short)0, acc1, false, false);
    }
    int n = nt * 16 + ml;
    float s = s1[n], t = t1[n];
#pragma unroll
    for (int r = 0; r < 8; ++r) {
      int m = r + 8 * hi;
      float y = acc0[r] * s + t;
      y = y > 0.f ? y : 0.01f * y;
      y1[wave][m * 256 + n] = f2bf(y);
      y = acc1[r] * s + t;
      y = y > 0.f ? y : 0.01f * y;
      y1[wave][(m + 16) * 256 + n] = f2bf(y);
    }
  }

  // layer 2: 256 -> 128
  for (int nt = 0; nt < 8; ++nt) {
    v8f acc0 = {}, acc1 = {};
    for (int kt = 0; kt < 8; ++kt) {
      int k0 = kt * 32 + 8 * hi;
      Frag a0, a1, bq;
      const unsigned short* ap = &y1[wave][ml * 256 + k0];
      a0.q[0] = *(const uint4*)ap;
      a0.q[1] = *(const uint4*)(ap + 16);
      a1.q[0] = *(const uint4*)(ap + 16 * 256);
      a1.q[1] = *(const uint4*)(ap + 16 * 256 + 16);
      const unsigned short* bp = w2b + (size_t)(kt * 32) * 128 + nt * 16 + ml;
      for (int s = 0; s < 16; ++s)
        ((unsigned short*)&bq)[s] = bp[(size_t)(8 * hi + (s & 7) + 16 * (s >> 3)) * 128];
      acc0 = __builtin_amdgcn_wmma_f32_16x16x32_bf16(
          false, a0.v, false, bq.v, (short)0, acc0, false, false);
      acc1 = __builtin_amdgcn_wmma_f32_16x16x32_bf16(
          false, a1.v, false, bq.v, (short)0, acc1, false, false);
    }
    int n = nt * 16 + ml;
    float s = s2[n], t = t2[n];
#pragma unroll
    for (int r = 0; r < 8; ++r) {
      int m = r + 8 * hi;
      float y = acc0[r] * s + t;
      y = y > 0.f ? y : 0.01f * y;
      y2[wave][m * 128 + n] = f2bf(y);
      y = acc1[r] * s + t;
      y = y > 0.f ? y : 0.01f * y;
      y2[wave][(m + 16) * 128 + n] = f2bf(y);
    }
  }

  // layer 3: 128 -> 64, write y to d_out
  for (int nt = 0; nt < 4; ++nt) {
    v8f acc0 = {}, acc1 = {};
    for (int kt = 0; kt < 4; ++kt) {
      int k0 = kt * 32 + 8 * hi;
      Frag a0, a1, bq;
      const unsigned short* ap = &y2[wave][ml * 128 + k0];
      a0.q[0] = *(const uint4*)ap;
      a0.q[1] = *(const uint4*)(ap + 16);
      a1.q[0] = *(const uint4*)(ap + 16 * 128);
      a1.q[1] = *(const uint4*)(ap + 16 * 128 + 16);
      const unsigned short* bp = w3b + (size_t)(kt * 32) * 64 + nt * 16 + ml;
      for (int s = 0; s < 16; ++s)
        ((unsigned short*)&bq)[s] = bp[(size_t)(8 * hi + (s & 7) + 16 * (s >> 3)) * 64];
      acc0 = __builtin_amdgcn_wmma_f32_16x16x32_bf16(
          false, a0.v, false, bq.v, (short)0, acc0, false, false);
      acc1 = __builtin_amdgcn_wmma_f32_16x16x32_bf16(
          false, a1.v, false, bq.v, (short)0, acc1, false, false);
    }
    int n = nt * 16 + ml;
    float s = s3[n], t = t3[n];
#pragma unroll
    for (int r = 0; r < 8; ++r) {
      int m = r + 8 * hi;
      float y = acc0[r] * s + t;
      y = y > 0.f ? y : 0.01f * y;
      out[(row0 + m) * 64 + n] = y;
      y = acc1[r] * s + t;
      y = y > 0.f ? y : 0.01f * y;
      out[(row0 + 16 + m) * 64 + n] = y;
    }
  }
}

// ---------------------------------------------------------------------------
extern "C" void kernel_launch(void* const* d_in, const int* in_sizes, int n_in,
                              void* d_out, int out_size, void* d_ws, size_t ws_size,
                              hipStream_t stream) {
  const float* x      = (const float*)d_in[0];
  const float* adj    = (const float*)d_in[1];
  const float* lgcn_w = (const float*)d_in[2];
  const float* lgcn_b = (const float*)d_in[3];
  const float* ca1w1  = (const float*)d_in[4];
  const float* ca1w2  = (const float*)d_in[5];
  const float* ca2w1  = (const float*)d_in[6];
  const float* ca2w2  = (const float*)d_in[7];
  const float* mw1    = (const float*)d_in[8];
  const float* mw2    = (const float*)d_in[9];
  const int*   idx16  = (const int*)d_in[10];
  const int*   idx7   = (const int*)d_in[11];
  const float* w1 = (const float*)d_in[12]; const float* b1 = (const float*)d_in[13];
  const float* w2 = (const float*)d_in[14]; const float* b2 = (const float*)d_in[15];
  const float* w3 = (const float*)d_in[16]; const float* b3 = (const float*)d_in[17];
  const float* g1 = (const float*)d_in[18]; const float* be1 = (const float*)d_in[19];
  const float* m1 = (const float*)d_in[20]; const float* v1  = (const float*)d_in[21];
  const float* g2 = (const float*)d_in[22]; const float* be2 = (const float*)d_in[23];
  const float* m2 = (const float*)d_in[24]; const float* v2  = (const float*)d_in[25];
  const float* g3 = (const float*)d_in[26]; const float* be3 = (const float*)d_in[27];
  const float* m3 = (const float*)d_in[28]; const float* v3  = (const float*)d_in[29];

  int B = in_sizes[0] / 310;
  char* ws = (char*)d_ws;

  unsigned short* wbig = (unsigned short*)(ws + 0);          // 320*320*2 = 204800
  unsigned short* w1b  = (unsigned short*)(ws + 204800);     // 736*256*2 = 376832
  unsigned short* w2b  = (unsigned short*)(ws + 581632);     // 256*128*2 = 65536
  unsigned short* w3b  = (unsigned short*)(ws + 647168);     // 128*64*2  = 16384
  float*          st   = (float*)(ws + 663552);              // 896 f32   = 3584
  // xb (B*320 bf16) and cat (B*736 bf16) overlap: xb is dead before cat is written
  unsigned short* xb   = (unsigned short*)(ws + 667136);
  unsigned short* cat  = (unsigned short*)(ws + 667136);
  float*          resf = (float*)(ws + 667136 + (size_t)B * 736 * 2);

  const int NW = 320*320 + 736*256 + 256*128 + 128*64 + 256 + 128 + 64;
  build_weights<<<(NW + 255) / 256, 256, 0, stream>>>(
      adj, lgcn_w, w1, b1, g1, be1, m1, v1, w2, b2, g2, be2, m2, v2,
      w3, b3, g3, be3, m3, v3, wbig, w1b, w2b, w3b, st);

  size_t nx = (size_t)B * 320;
  cvt_x<<<(unsigned)((nx + 255) / 256), 256, 0, stream>>>(x, xb, B);

  gemm_lgcn<<<B / 128, 128, 0, stream>>>(xb, wbig, lgcn_b, resf, B);

  prep_kernel<<<B / 32, 32, 32 * 344 * sizeof(float), stream>>>(
      resf, ca1w1, ca1w2, ca2w1, ca2w2, mw1, mw2, idx16, idx7,
      cat, (float*)d_out, B);

  mlp3<<<B / 64, 64, 0, stream>>>(cat, w1b, w2b, w3b, st, (float*)d_out, B);
}
